// W8A8BFP32OFP32Linear_71725953843873
// MI455X (gfx1250) — 8-bit matrix operand order repaired and checked by reading, compiled, not run on hardware
//
#include <hip/hip_runtime.h>
#include <stdint.h>

typedef int          v8i __attribute__((ext_vector_type(8)));
typedef int          v4i __attribute__((ext_vector_type(4)));
typedef int          v2i __attribute__((ext_vector_type(2)));
typedef float        v4f __attribute__((ext_vector_type(4)));
typedef unsigned int v4u __attribute__((ext_vector_type(4)));

#define M_TOT (4 * 4096)
#define N_TOT 4096
#define K_TOT 4096

#define KSTEP      128
#define WG_M       256
#define WG_N       128
#define ROW_STRIDE 144
#define A_BUF      (WG_M * ROW_STRIDE)
#define B_BUF      (WG_N * ROW_STRIDE)
#define SMEM_BYTES (2 * A_BUF + 2 * B_BUF)

__global__ __launch_bounds__(256) void pack_w_kernel(const int* __restrict__ wq,
                                                     int* __restrict__ wp) {
    int i = blockIdx.x * blockDim.x + threadIdx.x;
    const v4i* src = (const v4i*)wq;
    v4i v = src[i];
    wp[i] = (v[0] & 0xFF) | ((v[1] & 0xFF) << 8) |
            ((v[2] & 0xFF) << 16) | ((v[3] & 0xFF) << 24);
}

__global__ __launch_bounds__(256) void quant_x_kernel(const float* __restrict__ x,
                                                      int* __restrict__ xq,
                                                      float* __restrict__ ascale) {
    const int token = blockIdx.x;
    const int t = threadIdx.x;
    const v4f* r4 = (const v4f*)(x + (size_t)token * K_TOT);

    v4f v[4];
    float amax = 0.0f;
#pragma unroll
    for (int j = 0; j < 4; ++j) {
        v[j] = r4[t * 4 + j];
#pragma unroll
        for (int e = 0; e < 4; ++e) amax = fmaxf(amax, fabsf(v[j][e]));
    }
    __shared__ float red[256];
    red[t] = amax;
    __syncthreads();
#pragma unroll
    for (int s = 128; s > 0; s >>= 1) {
        if (t < s) red[t] = fmaxf(red[t], red[t + s]);
        __syncthreads();
    }
    const float scale = fmaxf(red[0] * (1.0f / 127.0f), 1e-8f);

    v4i out;
#pragma unroll
    for (int j = 0; j < 4; ++j) {
        int p = 0;
#pragma unroll
        for (int e = 0; e < 4; ++e) {
            int q = (int)rintf(v[j][e] / scale);
            q = q < -128 ? -128 : (q > 127 ? 127 : q);
            p |= (q & 0xFF) << (8 * e);
        }
        out[j] = p;
    }
    ((v4i*)(xq + (size_t)token * (K_TOT / 4)))[t] = out;
    if (t == 0) ascale[token] = scale;
}

__device__ __attribute__((always_inline)) void tdm_load_tile(const void* gptr,
                                                             unsigned lds_off,
                                                             int rows_total,
                                                             int tile_rows) {
    const unsigned long long ga = (unsigned long long)(uintptr_t)gptr;
    v4u g0;
    g0[0] = 1u;
    g0[1] = lds_off;
    g0[2] = (unsigned)(ga & 0xFFFFFFFFu);
    g0[3] = (unsigned)((ga >> 32) & 0x1FFFFFFu) |
            (2u << 30);

    v8i g1;
    g1[0] = (0  ) | (0 << 16  ) |
            (1 << 20  ) | (4 << 22  ) |
            (3 << 25  );
    g1[1] = (K_TOT & 0xFFFF) << 16;
    g1[2] = ((K_TOT >> 16) & 0xFFFF) | ((rows_total & 0xFFFF) << 16);
    g1[3] = ((rows_total >> 16) & 0xFFFF) | (KSTEP << 16);
    g1[4] = tile_rows & 0xFFFF;
    g1[5] = K_TOT;
    g1[6] = 0;
    g1[7] = 0;

    v4i gz = {0, 0, 0, 0};
#if defined(__clang_major__) && (__clang_major__ >= 23)
    v8i gz8 = {0, 0, 0, 0, 0, 0, 0, 0};
    __builtin_amdgcn_tensor_load_to_lds(g0, g1, gz, gz, gz8, 0);
#else
    __builtin_amdgcn_tensor_load_to_lds(g0, g1, gz, gz, 0);
#endif
}

__global__ __launch_bounds__(512, 1) void gemm_wmma_tdm_kernel(
    const char* __restrict__ xq, const char* __restrict__ wp,
    const float* __restrict__ ascale, const float* __restrict__ wscale,
    const float* __restrict__ bias, float* __restrict__ out) {
    extern __shared__ char smem[];
    const unsigned lds_base = (unsigned)(uintptr_t)smem;

    const int lane = threadIdx.x & 31;
    const int wave = threadIdx.x >> 5;
    const int col  = lane & 15;
    const int half = lane >> 4;
    const int wm = wave >> 1;
    const int wn = wave & 1;
    const int m_base = blockIdx.y * WG_M;
    const int n_base = blockIdx.x * WG_N;
    const int m_off = wm * 32;
    const int n_off = wn * 64;

    const char* Ag = xq + (size_t)m_base * K_TOT;
    const char* Bg = wp + (size_t)n_base * K_TOT;

    v8i acc[2][4];
#pragma unroll
    for (int mi = 0; mi < 2; ++mi)
#pragma unroll
        for (int ni = 0; ni < 4; ++ni)
#pragma unroll
            for (int r = 0; r < 8; ++r) acc[mi][ni][r] = 0;

    constexpr int NS = K_TOT / KSTEP;

    if (wave == 0) {
        tdm_load_tile(Ag, lds_base, M_TOT - m_base, WG_M);
        tdm_load_tile(Bg, lds_base + 2 * A_BUF, N_TOT - n_base, WG_N);
    }

    for (int ks = 0; ks < NS; ++ks) {
        if (wave == 0) {
            if (ks + 1 < NS) {
                const int kn = (ks + 1) * KSTEP;
                const unsigned nb = (unsigned)((ks + 1) & 1);
                tdm_load_tile(Ag + kn, lds_base + nb * A_BUF,
                              M_TOT - m_base, WG_M);
                tdm_load_tile(Bg + kn, lds_base + 2 * A_BUF + nb * B_BUF,
                              N_TOT - n_base, WG_N);
                __builtin_amdgcn_s_wait_tensorcnt(2);
            } else {
                __builtin_amdgcn_s_wait_tensorcnt(0);
            }
        }
        __syncthreads();

        const char* Ab = smem + (ks & 1) * A_BUF;
        const char* Bb = smem + 2 * A_BUF + (ks & 1) * B_BUF;
#pragma unroll
        for (int k64 = 0; k64 < KSTEP; k64 += 64) {
            v8i bfrag[4];
#pragma unroll
            for (int ni = 0; ni < 4; ++ni) {
                const char* bp = Bb + (n_off + ni * 16 + col) * ROW_STRIDE + k64 + 16 * half;
                v4i u0 = *(const v4i*)bp;
                v4i u1 = *(const v4i*)(bp + 32);
                bfrag[ni][0] = u0[0]; bfrag[ni][1] = u0[1];
                bfrag[ni][2] = u0[2]; bfrag[ni][3] = u0[3];
                bfrag[ni][4] = u1[0]; bfrag[ni][5] = u1[1];
                bfrag[ni][6] = u1[2]; bfrag[ni][7] = u1[3];
            }
#pragma unroll
            for (int mi = 0; mi < 2; ++mi) {
                const char* ap = Ab + (m_off + mi * 16 + col) * ROW_STRIDE + k64 + 16 * half;
                v2i t0 = *(const v2i*)ap;
                v2i t1 = *(const v2i*)(ap + 8);
                v2i t2 = *(const v2i*)(ap + 32);
                v2i t3 = *(const v2i*)(ap + 40);
                v8i a;
                a[0] = t0[0]; a[1] = t0[1]; a[2] = t1[0]; a[3] = t1[1];
                a[4] = t2[0]; a[5] = t2[1]; a[6] = t3[0]; a[7] = t3[1];
#pragma unroll
                for (int ni = 0; ni < 4; ++ni)
                    acc[mi][ni] = __builtin_amdgcn_wmma_i32_16x16x64_iu8(
                        true, a, true, bfrag[ni], acc[mi][ni], false, false);
            }
        }
        __syncthreads();
    }

    float as_[2][8];
#pragma unroll
    for (int mi = 0; mi < 2; ++mi)
#pragma unroll
        for (int r = 0; r < 8; ++r)
            as_[mi][r] = ascale[m_base + m_off + mi * 16 + r + 8 * half];

#pragma unroll
    for (int mi = 0; mi < 2; ++mi)
#pragma unroll
        for (int ni = 0; ni < 4; ++ni) {
            const int n = n_base + n_off + ni * 16 + col;
            const float wsc = wscale[n];
            const float bs = bias[n];
#pragma unroll
            for (int r = 0; r < 8; ++r) {
                const int m = m_base + m_off + mi * 16 + r + 8 * half;
                out[(size_t)m * N_TOT + n] = (float)acc[mi][ni][r] * as_[mi][r] * wsc + bs;
            }
        }
}

extern "C" void kernel_launch(void* const* d_in, const int* in_sizes, int n_in,
                              void* d_out, int out_size, void* d_ws, size_t ws_size,
                              hipStream_t stream) {
    const float* x      = (const float*)d_in[0];
    const int*   wq     = (const int*)d_in[1];
    const float* wscale = (const float*)d_in[2];
    const float* bias   = (const float*)d_in[3];
    float* out = (float*)d_out;

    char* wp = (char*)d_ws;
    char* xq = wp + (size_t)N_TOT * K_TOT;
    float* ascale = (float*)(xq + (size_t)M_TOT * K_TOT);
    (void)in_sizes; (void)n_in; (void)out_size; (void)ws_size;

    {
        const long long total4 = (long long)N_TOT * K_TOT / 4;
        pack_w_kernel<<<(unsigned)((total4 + 255) / 256), 256, 0, stream>>>(wq, (int*)wp);
    }
    quant_x_kernel<<<M_TOT, 256, 0, stream>>>(x, (int*)xq, ascale);
    {
        dim3 grid(N_TOT / WG_N, M_TOT / WG_M);
        gemm_wmma_tdm_kernel<<<grid, 512, SMEM_BYTES, stream>>>(
            xq, wp, ascale, wscale, bias, out);
    }
}
